// layer_32280974197147
// MI455X (gfx1250) — matrix operand order repaired and checked by reading, not compiled, not run on hardware
//
#include <hip/hip_runtime.h>
#include <math.h>

typedef __bf16 bf16;
typedef __attribute__((ext_vector_type(16))) __bf16 v16bf;
typedef __attribute__((ext_vector_type(8)))  __bf16 v8bf;
typedef __attribute__((ext_vector_type(8)))  float  v8f;

#define ENC    512
#define RNN    1024
#define G4     4096          // 4*RNN
#define SEQLEN 48
#define B      64
#define VOCAB  16000
#define VP1    16001         // VOCAB+1 (output logits)
#define VP1P   16016         // padded to multiple of 16
#define EMBROW 16002         // VOCAB+2 embedding rows

// ---------------------------------------------------------------------------
// Fragment loaders (CDNA5 WMMA 16x16x32 bf16, wave32).
// A (16x32, MxK): lane L<16 holds row M=L, K = k0..k0+15 (contiguous);
//                 lane L>=16 holds row M=L-16, K = k0+16..k0+31.
// B (32x16, KxN): lane L<16 holds col N=L, K = k0..k0+15 (contiguous);
//                 lane L>=16 holds col N=L-16, K = k0+16..k0+31.
// C/D (16x16 f32): lane L -> N = L&15; VGPR v -> M = v + (L>>4)*8.
// ---------------------------------------------------------------------------
__device__ __forceinline__ v16bf load_A_frag(const bf16* __restrict__ x,
                                             int ldx, int mTile, int k0) {
  const int lane = threadIdx.x & 31;
  const int m = lane & 15, hi = lane >> 4;
  const bf16* p = x + (size_t)(mTile + m) * ldx + k0 + hi * 16;
  v8bf lo = *(const v8bf*)(p);        // K = k0 + hi*16 .. +7
  v8bf hh = *(const v8bf*)(p + 8);    // K = k0 + hi*16 + 8 .. +15
  v16bf a;
#pragma unroll
  for (int i = 0; i < 8; ++i) { a[i] = lo[i]; a[8 + i] = hh[i]; }
  return a;
}

__device__ __forceinline__ v16bf load_B_frag(const bf16* __restrict__ w,
                                             int ldw, int nTile, int k0) {
  const int lane = threadIdx.x & 31;
  const int n = lane & 15, hi = lane >> 4;
  const bf16* p = w + (size_t)(nTile + n) * ldw + k0 + hi * 16;
  return *(const v16bf*)(p);          // 16 contiguous bf16 = 32B aligned
}

// Store one 16x16 f32 D tile (C/D VGPR layout) with a per-column bias.
__device__ __forceinline__ void store_D_tile(float* __restrict__ dst, int ldd,
                                             int mTile, int nTile,
                                             const v8f& acc, float bias) {
  const int lane = threadIdx.x & 31;
  const int n = nTile + (lane & 15);
  const int hi = lane >> 4;
#pragma unroll
  for (int v = 0; v < 8; ++v) {
    const int m = mTile + v + hi * 8;
    dst[(size_t)m * ldd + n] = acc[v] + bias;
  }
}

// ---------------------------------------------------------------------------
// One-time (per call) weight conversion f32 -> bf16, with W_out padded to
// 16016 rows (zeros) so the logits GEMM needs no N-edge handling.
// ---------------------------------------------------------------------------
__global__ void cvt_weights_kernel(const float* __restrict__ emb,
                                   const float* __restrict__ Wi,
                                   const float* __restrict__ Wh,
                                   const float* __restrict__ Wo,
                                   const float* __restrict__ bo,
                                   bf16* __restrict__ emb_bf,
                                   bf16* __restrict__ Wi_bf,
                                   bf16* __restrict__ Wh_bf,
                                   bf16* __restrict__ Wo_bf,
                                   float* __restrict__ bo_pad) {
  const size_t tid = (size_t)blockIdx.x * blockDim.x + threadIdx.x;
  const size_t stride = (size_t)gridDim.x * blockDim.x;
  for (size_t i = tid; i < (size_t)EMBROW * ENC; i += stride) emb_bf[i] = (bf16)emb[i];
  for (size_t i = tid; i < (size_t)G4 * ENC;    i += stride) Wi_bf[i]  = (bf16)Wi[i];
  for (size_t i = tid; i < (size_t)G4 * RNN;    i += stride) Wh_bf[i]  = (bf16)Wh[i];
  for (size_t i = tid; i < (size_t)VP1P * RNN;  i += stride)
    Wo_bf[i] = (i < (size_t)VP1 * RNN) ? (bf16)Wo[i] : (bf16)0.0f;
  for (size_t i = tid; i < (size_t)VP1P;        i += stride)
    bo_pad[i] = (i < (size_t)VP1) ? bo[i] : 0.0f;
}

__global__ void init_state_kernel(float* __restrict__ c, float* __restrict__ h,
                                  bf16* __restrict__ h_bf) {
  const int i = blockIdx.x * blockDim.x + threadIdx.x;
  if (i < B * RNN) { c[i] = 0.0f; h[i] = 0.0f; h_bf[i] = (bf16)0.0f; }
}

// ---------------------------------------------------------------------------
// Per-step input gather: x_bf[64,512] and the padding-skip flag.
// ---------------------------------------------------------------------------
__global__ void prep_x_kernel(const float* __restrict__ imgs,
                              const bf16* __restrict__ emb_bf,
                              const int* __restrict__ seq,
                              int t, bf16* __restrict__ x,
                              int* __restrict__ skip) {
  const int idx = blockIdx.x * blockDim.x + threadIdx.x;
  if (idx < B * ENC) {
    const int b = idx >> 9, k = idx & (ENC - 1);
    if (t == 0)      x[idx] = (bf16)imgs[idx];
    else if (t == 1) x[idx] = emb_bf[(size_t)(VOCAB + 1) * ENC + k];
    else {
      const int tok = seq[(t - 2) * B + b];
      x[idx] = emb_bf[(size_t)tok * ENC + k];
    }
  }
  if (blockIdx.x == 0 && threadIdx.x == 0) {
    int s = 1;  // non-zero => don't skip
    if (t >= 2) {
      s = 0;
      for (int b = 0; b < B; ++b) s += seq[(t - 2) * B + b];
    }
    *skip = (s == 0) ? 1 : 0;
  }
}

// ---------------------------------------------------------------------------
// gates[64,4096] = x[64,512] @ Wi^T + h[64,1024] @ Wh^T + (b_i2h + b_h2h)
// Each wave owns one N-tile and computes ALL FOUR M-tiles (M = 64), so every
// B (weight) fragment is loaded once and reused by 4 WMMAs -> 4x less L2
// weight traffic. 4 waves/block. grid = (64, 1), block = 128.
// ---------------------------------------------------------------------------
__global__ void gates_gemm_kernel(const bf16* __restrict__ x,
                                  const bf16* __restrict__ hprev,
                                  const bf16* __restrict__ Wi,
                                  const bf16* __restrict__ Wh,
                                  const float* __restrict__ bi,
                                  const float* __restrict__ bh,
                                  float* __restrict__ gates) {
  const int wave  = threadIdx.x >> 5;
  const int nTile = (blockIdx.x * 4 + wave) * 16;   // 0..4080
  v8f acc0 = {}, acc1 = {}, acc2 = {}, acc3 = {};
#pragma unroll 2
  for (int k0 = 0; k0 < ENC; k0 += 32) {
    v16bf b  = load_B_frag(Wi, ENC, nTile, k0);
    v16bf a0 = load_A_frag(x, ENC,  0, k0);
    v16bf a1 = load_A_frag(x, ENC, 16, k0);
    v16bf a2 = load_A_frag(x, ENC, 32, k0);
    v16bf a3 = load_A_frag(x, ENC, 48, k0);
    acc0 = __builtin_amdgcn_wmma_f32_16x16x32_bf16(false, a0, false, b, (short)0, acc0, false, false);
    acc1 = __builtin_amdgcn_wmma_f32_16x16x32_bf16(false, a1, false, b, (short)0, acc1, false, false);
    acc2 = __builtin_amdgcn_wmma_f32_16x16x32_bf16(false, a2, false, b, (short)0, acc2, false, false);
    acc3 = __builtin_amdgcn_wmma_f32_16x16x32_bf16(false, a3, false, b, (short)0, acc3, false, false);
  }
#pragma unroll 2
  for (int k0 = 0; k0 < RNN; k0 += 32) {
    v16bf b  = load_B_frag(Wh, RNN, nTile, k0);
    v16bf a0 = load_A_frag(hprev, RNN,  0, k0);
    v16bf a1 = load_A_frag(hprev, RNN, 16, k0);
    v16bf a2 = load_A_frag(hprev, RNN, 32, k0);
    v16bf a3 = load_A_frag(hprev, RNN, 48, k0);
    acc0 = __builtin_amdgcn_wmma_f32_16x16x32_bf16(false, a0, false, b, (short)0, acc0, false, false);
    acc1 = __builtin_amdgcn_wmma_f32_16x16x32_bf16(false, a1, false, b, (short)0, acc1, false, false);
    acc2 = __builtin_amdgcn_wmma_f32_16x16x32_bf16(false, a2, false, b, (short)0, acc2, false, false);
    acc3 = __builtin_amdgcn_wmma_f32_16x16x32_bf16(false, a3, false, b, (short)0, acc3, false, false);
  }
  const int n = nTile + ((threadIdx.x & 31) & 15);
  const float bias = bi[n] + bh[n];
  store_D_tile(gates, G4,  0, nTile, acc0, bias);
  store_D_tile(gates, G4, 16, nTile, acc1, bias);
  store_D_tile(gates, G4, 32, nTile, acc2, bias);
  store_D_tile(gates, G4, 48, nTile, acc3, bias);
}

// ---------------------------------------------------------------------------
// LSTM cell pointwise update. Split order i,f,o,g (matches jnp.split).
// ---------------------------------------------------------------------------
__device__ __forceinline__ float sigmoidf(float v) { return 1.0f / (1.0f + __expf(-v)); }

__global__ void lstm_cell_kernel(const float* __restrict__ gates,
                                 const int* __restrict__ skip,
                                 float* __restrict__ c, float* __restrict__ h,
                                 bf16* __restrict__ h_bf) {
  if (*skip) return;  // keep previous c/h/h_bf
  const int idx = blockIdx.x * blockDim.x + threadIdx.x;
  if (idx >= B * RNN) return;
  const int b = idx >> 10, n = idx & (RNN - 1);
  const float* g = gates + (size_t)b * G4;
  const float ig = sigmoidf(g[n]);
  const float fg = sigmoidf(g[RNN + n]);
  const float og = sigmoidf(g[2 * RNN + n]);
  const float gg = tanhf(g[3 * RNN + n]);
  const float cn = fg * c[idx] + ig * gg;
  const float hn = og * tanhf(cn);
  c[idx] = cn;
  h[idx] = hn;
  h_bf[idx] = (bf16)hn;
}

// ---------------------------------------------------------------------------
// logits[64,16016] = h[64,1024] @ Wo^T + b_out (padded cols are zero weight).
// Each wave owns one N-tile, computes all 4 M-tiles: W_out streamed from L2
// exactly ONCE per step. grid = (251, 1), block = 128; 1001 real N-tiles.
// ---------------------------------------------------------------------------
__global__ void logits_gemm_kernel(const bf16* __restrict__ h_bf,
                                   const bf16* __restrict__ Wo,
                                   const float* __restrict__ bo,
                                   float* __restrict__ logits) {
  const int wave = threadIdx.x >> 5;
  const int tIdx = blockIdx.x * 4 + wave;
  if (tIdx >= VP1P / 16) return;
  const int nTile = tIdx * 16;
  v8f acc0 = {}, acc1 = {}, acc2 = {}, acc3 = {};
#pragma unroll 2
  for (int k0 = 0; k0 < RNN; k0 += 32) {
    v16bf b  = load_B_frag(Wo, RNN, nTile, k0);
    v16bf a0 = load_A_frag(h_bf, RNN,  0, k0);
    v16bf a1 = load_A_frag(h_bf, RNN, 16, k0);
    v16bf a2 = load_A_frag(h_bf, RNN, 32, k0);
    v16bf a3 = load_A_frag(h_bf, RNN, 48, k0);
    acc0 = __builtin_amdgcn_wmma_f32_16x16x32_bf16(false, a0, false, b, (short)0, acc0, false, false);
    acc1 = __builtin_amdgcn_wmma_f32_16x16x32_bf16(false, a1, false, b, (short)0, acc1, false, false);
    acc2 = __builtin_amdgcn_wmma_f32_16x16x32_bf16(false, a2, false, b, (short)0, acc2, false, false);
    acc3 = __builtin_amdgcn_wmma_f32_16x16x32_bf16(false, a3, false, b, (short)0, acc3, false, false);
  }
  const int n = nTile + ((threadIdx.x & 31) & 15);
  const float bias = bo[n];
  store_D_tile(logits, VP1P,  0, nTile, acc0, bias);
  store_D_tile(logits, VP1P, 16, nTile, acc1, bias);
  store_D_tile(logits, VP1P, 32, nTile, acc2, bias);
  store_D_tile(logits, VP1P, 48, nTile, acc3, bias);
}

// ---------------------------------------------------------------------------
// Row-wise log-softmax over 16001 entries; zeros on skipped steps.
// One block (256 threads) per batch row.
// ---------------------------------------------------------------------------
__global__ void logsoftmax_kernel(const float* __restrict__ logits,
                                  const int* __restrict__ skip,
                                  float* __restrict__ out) {
  const int b = blockIdx.x;
  const float* row = logits + (size_t)b * VP1P;
  float* orow = out + (size_t)b * VP1;
  __shared__ float red[256];
  if (*skip) {  // uniform branch across the block
    for (int i = threadIdx.x; i < VP1; i += 256) orow[i] = 0.0f;
    return;
  }
  float m = -3.402823466e38f;
  for (int i = threadIdx.x; i < VP1; i += 256) m = fmaxf(m, row[i]);
  red[threadIdx.x] = m;
  __syncthreads();
  for (int s = 128; s > 0; s >>= 1) {
    if (threadIdx.x < s) red[threadIdx.x] = fmaxf(red[threadIdx.x], red[threadIdx.x + s]);
    __syncthreads();
  }
  m = red[0];
  __syncthreads();
  float sum = 0.0f;
  for (int i = threadIdx.x; i < VP1; i += 256) sum += __expf(row[i] - m);
  red[threadIdx.x] = sum;
  __syncthreads();
  for (int s = 128; s > 0; s >>= 1) {
    if (threadIdx.x < s) red[threadIdx.x] += red[threadIdx.x + s];
    __syncthreads();
  }
  const float lse = m + __logf(red[0]);
  for (int i = threadIdx.x; i < VP1; i += 256) orow[i] = row[i] - lse;
}

// ---------------------------------------------------------------------------
// Host-side orchestration (graph-capture safe: only async launches on stream)
// ---------------------------------------------------------------------------
extern "C" void kernel_launch(void* const* d_in, const int* in_sizes, int n_in,
                              void* d_out, int out_size, void* d_ws, size_t ws_size,
                              hipStream_t stream) {
  const float* imgs  = (const float*)d_in[0];
  const int*   seq   = (const int*)  d_in[1];
  const float* emb   = (const float*)d_in[2];
  const float* W_i2h = (const float*)d_in[3];
  const float* b_i2h = (const float*)d_in[4];
  const float* W_h2h = (const float*)d_in[5];
  const float* b_h2h = (const float*)d_in[6];
  const float* W_out = (const float*)d_in[7];
  const float* b_out = (const float*)d_in[8];
  float* out = (float*)d_out;

  // workspace carve-up (256B aligned slices)
  char* ws = (char*)d_ws;
  size_t off = 0;
  auto alloc = [&](size_t bytes) -> char* {
    char* p = ws + off;
    off = (off + bytes + 255) & ~(size_t)255;
    return p;
  };
  bf16*  emb_bf = (bf16*) alloc((size_t)EMBROW * ENC * sizeof(bf16));
  bf16*  Wi_bf  = (bf16*) alloc((size_t)G4 * ENC * sizeof(bf16));
  bf16*  Wh_bf  = (bf16*) alloc((size_t)G4 * RNN * sizeof(bf16));
  bf16*  Wo_bf  = (bf16*) alloc((size_t)VP1P * RNN * sizeof(bf16));
  float* bo_pad = (float*)alloc((size_t)VP1P * sizeof(float));
  bf16*  x_bf   = (bf16*) alloc((size_t)B * ENC * sizeof(bf16));
  bf16*  h_bf   = (bf16*) alloc((size_t)B * RNN * sizeof(bf16));
  float* c_st   = (float*)alloc((size_t)B * RNN * sizeof(float));
  float* h_st   = (float*)alloc((size_t)B * RNN * sizeof(float));
  float* gates  = (float*)alloc((size_t)B * G4 * sizeof(float));
  float* logits = (float*)alloc((size_t)B * VP1P * sizeof(float));
  int*   skip   = (int*)  alloc(256);

  // one-time per call: bf16 weights + zero state
  cvt_weights_kernel<<<2048, 256, 0, stream>>>(emb, W_i2h, W_h2h, W_out, b_out,
                                               emb_bf, Wi_bf, Wh_bf, Wo_bf, bo_pad);
  init_state_kernel<<<(B * RNN + 255) / 256, 256, 0, stream>>>(c_st, h_st, h_bf);

  const int TOTAL_T = SEQLEN + 2;  // 50 timesteps
  for (int t = 0; t < TOTAL_T; ++t) {
    prep_x_kernel<<<(B * ENC + 255) / 256, 256, 0, stream>>>(imgs, emb_bf, seq, t,
                                                             x_bf, skip);
    gates_gemm_kernel<<<G4 / 64, 128, 0, stream>>>(
        x_bf, h_bf, Wi_bf, Wh_bf, b_i2h, b_h2h, gates);
    lstm_cell_kernel<<<(B * RNN + 255) / 256, 256, 0, stream>>>(gates, skip,
                                                                c_st, h_st, h_bf);
    logits_gemm_kernel<<<(VP1P / 16 + 3) / 4, 128, 0, stream>>>(
        h_bf, Wo_bf, bo_pad, logits);
    logsoftmax_kernel<<<B, 256, 0, stream>>>(logits, skip,
                                             out + (size_t)t * B * VP1);
  }
}
